// KDABlock_41764261986321
// MI455X (gfx1250) — matrix operand order repaired and checked by reading, not compiled, not run on hardware
//
#include <hip/hip_runtime.h>
#include <hip/hip_bf16.h>

typedef _Float16 f16;
typedef _Float16 v16h __attribute__((ext_vector_type(16)));
typedef _Float16 h8v  __attribute__((ext_vector_type(8)));
typedef float    v8f  __attribute__((ext_vector_type(8)));
typedef int      i4v  __attribute__((ext_vector_type(4)));

typedef __attribute__((address_space(1))) i4v* gi4p;   // global int4*
typedef __attribute__((address_space(3))) i4v* li4p;   // LDS int4*

#define EPSF   1e-6f
#define DECAYF 0.9f
#define THRF   0.5f

#if __has_builtin(__builtin_amdgcn_global_load_async_to_lds_b128) && \
    __has_builtin(__builtin_amdgcn_s_wait_asynccnt)
#define HAS_ASYNC_LDS 1
#else
#define HAS_ASYNC_LDS 0
#endif

__device__ __forceinline__ float sigm(float x) { return 1.0f / (1.0f + __expf(-x)); }

// ---------------- elementwise cast f32 -> f16 ----------------
__global__ void cast16_kernel(const float* __restrict__ in, f16* __restrict__ out, int n) {
    int i = blockIdx.x * blockDim.x + threadIdx.x;
    if (i < n) out[i] = (f16)in[i];
}

// ---------------- weight prep: W[K,N] f32 -> WT[Npad,K] f16 (transposed, zero-padded rows) ----
__global__ void transpose_cast_kernel(const float* __restrict__ W, f16* __restrict__ WT,
                                      int K, int N, int Npad) {
    long long idx = (long long)blockIdx.x * blockDim.x + threadIdx.x;
    if (idx >= (long long)Npad * K) return;
    int n = (int)(idx % Npad);      // consecutive threads -> consecutive n: coalesced W reads
    int k = (int)(idx / Npad);
    float v = (n < N) ? W[(size_t)k * N + n] : 0.f;
    WT[(size_t)n * K + k] = (f16)v;
}

// ---------------- RMS norm (per row over D), optional f32/f16 outputs ----------------
__global__ void rmsnorm_kernel(const float* __restrict__ x, const float* __restrict__ w,
                               float* __restrict__ o32, f16* __restrict__ o16, int D) {
    __shared__ float red[256];
    int row = blockIdx.x, tid = threadIdx.x;
    const float* xr = x + (size_t)row * D;
    float ss = 0.f;
    for (int i = tid; i < D; i += 256) { float v = xr[i]; ss += v * v; }
    red[tid] = ss; __syncthreads();
    for (int s = 128; s > 0; s >>= 1) { if (tid < s) red[tid] += red[tid + s]; __syncthreads(); }
    float scale = rsqrtf(red[0] / (float)D + EPSF);
    for (int i = tid; i < D; i += 256) {
        float v = xr[i] * scale * w[i];
        if (o32) o32[(size_t)row * D + i] = v;
        if (o16) o16[(size_t)row * D + i] = (f16)v;
    }
}

// ---------------- causal depthwise conv (KW=4) + SiLU, writes f16 ----------------
__global__ void conv_silu_kernel(const float* __restrict__ hin, const float* __restrict__ cw,
                                 const float* __restrict__ cb, f16* __restrict__ hout,
                                 int T, int D) {
    int row = blockIdx.x;            // b*T + t
    int t = row % T;
    int tid = threadIdx.x;
    for (int d = tid; d < D; d += 256) {
        float acc = cb[d];
#pragma unroll
        for (int j = 0; j < 4; ++j) {
            int tt = t - 3 + j;
            if (tt >= 0) acc += hin[(size_t)(row - 3 + j) * D + d] * cw[d * 4 + j];
        }
        float o = acc * sigm(acc);
        hout[(size_t)row * D + d] = (f16)o;
    }
}

// ---------------- per-head L2 normalize (in place), HD=512, dk=64 ----------------
__global__ void l2norm_head_kernel(float* __restrict__ q) {
    __shared__ float red[512];
    int row = blockIdx.x, tid = threadIdx.x;        // block = 512 threads
    float v = q[(size_t)row * 512 + tid];
    red[tid] = v * v; __syncthreads();
    for (int s = 32; s > 0; s >>= 1) { if ((tid & 63) < s) red[tid] += red[tid + s]; __syncthreads(); }
    float nrm = sqrtf(red[tid & ~63]);
    q[(size_t)row * 512 + tid] = v / (nrm + 1e-6f);
}

// ---------------- per-head RMS norm * head_norm_w -> f16 ----------------
__global__ void headrms_kernel(const float* __restrict__ y, const float* __restrict__ hw,
                               f16* __restrict__ o16) {
    __shared__ float red[512];
    int row = blockIdx.x, tid = threadIdx.x;
    float v = y[(size_t)row * 512 + tid];
    red[tid] = v * v; __syncthreads();
    for (int s = 32; s > 0; s >>= 1) { if ((tid & 63) < s) red[tid] += red[tid + s]; __syncthreads(); }
    float mean = red[tid & ~63] * (1.0f / 64.0f);
    o16[(size_t)row * 512 + tid] = (f16)(v * rsqrtf(mean + EPSF) * hw[tid]);
}

// ---------------- sequential LIF + gated delta-rule scan; one block per (b,h) ----------------
__global__ void scan_kernel(const float* __restrict__ qf, const float* __restrict__ kf,
                            const float* __restrict__ vf, const float* __restrict__ drv,
                            const float* __restrict__ ab, const float* __restrict__ bb,
                            const float* __restrict__ asp, const float* __restrict__ bsp,
                            float* __restrict__ y, int T, int H) {
    __shared__ float qs[64], ks[64], vs[64], als[64], spk[64], mem[64], red[256];
    __shared__ float beta_sh;
    __shared__ int active_sh;
    int bid = blockIdx.x;
    int b = bid / H, h = bid % H;
    int tid = threadIdx.x;
    int v = tid & 63, k0 = (tid >> 6) * 16;     // 4 k-groups of 16
    float S[16];
#pragma unroll
    for (int i = 0; i < 16; ++i) S[i] = 0.f;
    if (tid < 64) mem[tid] = 0.f;
    __syncthreads();

    for (int t = 0; t < T; ++t) {
        size_t base = (size_t)(b * T + t) * 512 + (size_t)h * 64;
        if (tid < 64) {
            int kk = tid;
            float m = DECAYF * mem[kk] + drv[base + kk];
            float s = (m > THRF) ? 1.f : 0.f;   // forward value of straight-through spike
            mem[kk] = m - s * THRF;
            spk[kk] = s;
            qs[kk] = qf[base + kk];
            ks[kk] = kf[base + kk];
            vs[kk] = vf[base + kk];
            als[kk] = sigm(ab[base + kk] + asp[h * 64 + kk] * s);
            red[kk] = s;
            red[64 + kk] = s * bsp[h * 64 + kk];
        }
        __syncthreads();
        for (int s2 = 32; s2 > 0; s2 >>= 1) {
            if (tid < s2) { red[tid] += red[tid + s2]; red[64 + tid] += red[64 + tid + s2]; }
            __syncthreads();
        }
        if (tid == 0) {
            int act = red[0] > 0.f;
            active_sh = act;
            float bt = bb[(size_t)(b * T + t) * H + h];
            beta_sh = act ? sigm(bt + red[64]) : 0.f;
        }
        __syncthreads();
        int act = active_sh;
        float partial = 0.f;
#pragma unroll
        for (int i = 0; i < 16; ++i) {
            float a = act ? als[k0 + i] : 1.0f;
            S[i] *= a;
            partial += ks[k0 + i] * S[i];
        }
        red[tid] = partial; __syncthreads();
        float pred = red[v] + red[64 + v] + red[128 + v] + red[192 + v];
        float delta = beta_sh * (vs[v] - pred);
        float outp = 0.f;
#pragma unroll
        for (int i = 0; i < 16; ++i) { S[i] += ks[k0 + i] * delta; outp += qs[k0 + i] * S[i]; }
        __syncthreads();
        red[tid] = outp; __syncthreads();
        if (tid < 64) y[base + tid] = red[tid] + red[64 + tid] + red[128 + tid] + red[192 + tid];
        __syncthreads();
    }
}

// ---------------- WMMA GEMM: out = act(A[f16] @ WT^T + bias) (*mul) (+res) ----------------
// A: [M,K] f16 row-major. WT: [Npad>=ceil(N/128)*128, K] f16 (pre-transposed weights).
// Block tile 128x128, K-tile 32, double-buffered LDS, async global->LDS copies when available.
// 8 wave32 (4 along M x 2 along N), each wave 2x4 16x16 frags -> v_wmma_f32_16x16x32_f16.
__global__ __launch_bounds__(256) void gemm_wmma_kernel(
    const f16* __restrict__ A, const f16* __restrict__ WT,
    const float* __restrict__ bias, const float* __restrict__ mul,
    const float* __restrict__ res, float* __restrict__ out32,
    f16* __restrict__ out16, int M, int N, int K, int act) {
    __shared__ f16 Asb[2][128][40];   // [buf][m][k], stride 40 keeps 16B alignment
    __shared__ f16 Bsb[2][128][40];   // [buf][n][k]
    int tid = threadIdx.x;
    int lane = tid & 31, w = tid >> 5;
    int wm = w & 3, wn = w >> 2;
    int rowBase = blockIdx.y * 128;
    int nBase = blockIdx.x * 128;
    v8f acc[2][4] = {};
    (void)M;

    // copy one 128x32 K-tile of A and of WT into LDS buffer bufi (4 x 16B per thread)
    auto issue_tile = [&](int kk0, int bufi) {
#pragma unroll
        for (int it = 0; it < 2; ++it) {
            int idx = tid + it * 256;            // 512 slots of 8 halves each
            int rr = idx >> 2, c8 = (idx & 3) * 8;
            const f16* ga = A + (size_t)(rowBase + rr) * K + (kk0 + c8);
            const f16* gb = WT + (size_t)(nBase + rr) * K + (kk0 + c8);
#if HAS_ASYNC_LDS
            __builtin_amdgcn_global_load_async_to_lds_b128(
                (gi4p)(void*)ga, (li4p)(void*)&Asb[bufi][rr][c8], 0, 0);
            __builtin_amdgcn_global_load_async_to_lds_b128(
                (gi4p)(void*)gb, (li4p)(void*)&Bsb[bufi][rr][c8], 0, 0);
#else
            *(h8v*)&Asb[bufi][rr][c8] = *(const h8v*)ga;
            *(h8v*)&Bsb[bufi][rr][c8] = *(const h8v*)gb;
#endif
        }
    };

    issue_tile(0, 0);
    int buf = 0;
    for (int k0 = 0; k0 < K; k0 += 32, buf ^= 1) {
        bool more = (k0 + 32 < K);
        if (more) issue_tile(k0 + 32, buf ^ 1);   // prefetch next tile into other buffer
#if HAS_ASYNC_LDS
        // 4 async instrs per tile per wave; in-order retire => <=4 outstanding means tile k0 done
        if (more) __builtin_amdgcn_s_wait_asynccnt(4);
        else      __builtin_amdgcn_s_wait_asynccnt(0);
#endif
        if (k0 + 64 < K) {                        // L2 prefetch two tiles ahead
            if (tid < 32)
                __builtin_prefetch(A + (size_t)(rowBase + tid * 4) * K + (k0 + 64), 0, 1);
            else if (tid < 64)
                __builtin_prefetch(WT + (size_t)(nBase + (tid - 32) * 4) * K + (k0 + 64), 0, 1);
        }
        __syncthreads();

        int l15 = lane & 15;
        int kbA = (lane < 16) ? 0 : 8;   // A frag: lanes<16 K{0-7,16-23}; lanes>=16 K{8-15,24-31}
        int kbB = (lane < 16) ? 0 : 8;   // B frag: lanes<16 K{0-7,16-23}; lanes>=16 K{8-15,24-31}
        v16h afr[2], bfr[4];
#pragma unroll
        for (int mi = 0; mi < 2; ++mi) {
            const f16* p = &Asb[buf][wm * 32 + mi * 16 + l15][kbA];
            h8v lo = *(const h8v*)p;
            h8v hi = *(const h8v*)(p + 16);
            afr[mi] = __builtin_shufflevector(lo, hi, 0, 1, 2, 3, 4, 5, 6, 7, 8, 9, 10, 11, 12, 13, 14, 15);
        }
#pragma unroll
        for (int ni = 0; ni < 4; ++ni) {
            const f16* p = &Bsb[buf][wn * 64 + ni * 16 + l15][kbB];
            h8v lo = *(const h8v*)p;
            h8v hi = *(const h8v*)(p + 16);
            bfr[ni] = __builtin_shufflevector(lo, hi, 0, 1, 2, 3, 4, 5, 6, 7, 8, 9, 10, 11, 12, 13, 14, 15);
        }
#pragma unroll
        for (int mi = 0; mi < 2; ++mi)
#pragma unroll
            for (int ni = 0; ni < 4; ++ni)
                acc[mi][ni] = __builtin_amdgcn_wmma_f32_16x16x32_f16(
                    false, afr[mi], false, bfr[ni], (short)0, acc[mi][ni], false, false);
        __syncthreads();
    }

    // epilogue: bias -> act -> *mul -> +res -> store f32/f16
    int l15 = lane & 15;
#pragma unroll
    for (int mi = 0; mi < 2; ++mi) {
        int rb = rowBase + wm * 32 + mi * 16 + ((lane < 16) ? 0 : 8);
#pragma unroll
        for (int ni = 0; ni < 4; ++ni) {
            int cc = nBase + wn * 64 + ni * 16 + l15;
            if (cc < N) {
#pragma unroll
                for (int j = 0; j < 8; ++j) {
                    int rr = rb + j;
                    float val = acc[mi][ni][j];
                    if (bias) val += bias[cc];
                    if (act == 1) val = val * sigm(val);
                    else if (act == 2) val = sigm(val);
                    size_t o = (size_t)rr * N + cc;
                    if (mul) val *= mul[o];
                    if (res) val += res[o];
                    if (out32) out32[o] = val;
                    if (out16) out16[o] = (f16)val;
                }
            }
        }
    }
}

extern "C" void kernel_launch(void* const* d_in, const int* in_sizes, int n_in,
                              void* d_out, int out_size, void* d_ws, size_t ws_size,
                              hipStream_t stream) {
    (void)in_sizes; (void)n_in; (void)out_size; (void)ws_size;
    const int Bb = 2, T = 2048, D = 1024, H = 8, dk = 64, r = 64, dff = 4096;
    const int M = Bb * T;        // 4096
    const int HD = H * dk;       // 512
    (void)dk;

    const float* x     = (const float*)d_in[0];
    const float* nw    = (const float*)d_in[1];
    const float* cw    = (const float*)d_in[2];
    const float* cb    = (const float*)d_in[3];
    const float* Wq    = (const float*)d_in[4];
    const float* Wk    = (const float*)d_in[5];
    const float* Wv    = (const float*)d_in[6];
    const float* Wo    = (const float*)d_in[7];
    const float* Wsp   = (const float*)d_in[8];
    const float* Wau   = (const float*)d_in[9];
    const float* bau   = (const float*)d_in[10];
    const float* Wad   = (const float*)d_in[11];
    const float* bad_  = (const float*)d_in[12];
    const float* asp   = (const float*)d_in[13];
    const float* Wbeta = (const float*)d_in[14];
    const float* bbeta = (const float*)d_in[15];
    const float* bsp   = (const float*)d_in[16];
    const float* hnw   = (const float*)d_in[17];
    const float* Wu1   = (const float*)d_in[18];
    const float* bu1   = (const float*)d_in[19];
    const float* Wu2   = (const float*)d_in[20];
    const float* bu2   = (const float*)d_in[21];
    const float* ffw   = (const float*)d_in[22];
    const float* Wff1  = (const float*)d_in[23];
    const float* Wff3  = (const float*)d_in[24];
    const float* Wff2  = (const float*)d_in[25];
    float* out = (float*)d_out;

    char* ws = (char*)d_ws;
    size_t off = 0;
    auto alloc = [&](size_t bytes) -> void* {
        off = (off + 255) & ~(size_t)255;
        void* p = ws + off;
        off += bytes;
        return p;
    };
    auto npad = [](int N) { return ((N + 127) / 128) * 128; };

    // activations
    float* rms_tmp = (float*)alloc((size_t)M * D * 4);
    f16*   h16     = (f16*)alloc((size_t)M * D * 2);
    f16*   x16     = (f16*)alloc((size_t)M * D * 2);
    float* qf   = (float*)alloc((size_t)M * HD * 4);
    float* kf   = (float*)alloc((size_t)M * HD * 4);
    float* vf   = (float*)alloc((size_t)M * HD * 4);
    float* dr   = (float*)alloc((size_t)M * HD * 4);
    float* alb  = (float*)alloc((size_t)M * HD * 4);
    f16*   u16  = (f16*)alloc((size_t)M * r * 2);
    float* btb  = (float*)alloc((size_t)M * H * 4);
    float* ysc  = (float*)alloc((size_t)M * HD * 4);
    f16*   yn16 = (f16*)alloc((size_t)M * HD * 2);
    f16*   g16  = (f16*)alloc((size_t)M * r * 2);
    float* gate = (float*)alloc((size_t)M * D * 4);
    float* y2   = (float*)alloc((size_t)M * D * 4);
    f16*   z16  = (f16*)alloc((size_t)M * D * 2);
    float* f1   = (float*)alloc((size_t)M * dff * 4);
    f16*   p16  = (f16*)alloc((size_t)M * dff * 2);

    // pre-transposed f16 weights [Npad, K]
    auto prep = [&](const float* W_, int K, int N) -> f16* {
        int Np = npad(N);
        f16* WT = (f16*)alloc((size_t)Np * K * 2);
        long long tot = (long long)Np * K;
        transpose_cast_kernel<<<(int)((tot + 255) / 256), 256, 0, stream>>>(W_, WT, K, N, Np);
        return WT;
    };
    f16* WqT   = prep(Wq,   D, HD);
    f16* WkT   = prep(Wk,   D, HD);
    f16* WvT   = prep(Wv,   D, HD);
    f16* WspT  = prep(Wsp,  D, HD);
    f16* WauT  = prep(Wau,  D, r);
    f16* WadT  = prep(Wad,  r, HD);
    f16* WbT   = prep(Wbeta, D, H);
    f16* Wu1T  = prep(Wu1,  D, r);
    f16* Wu2T  = prep(Wu2,  r, D);
    f16* WoT   = prep(Wo,   HD, D);
    f16* Wff1T = prep(Wff1, D, dff);
    f16* Wff3T = prep(Wff3, D, dff);
    f16* Wff2T = prep(Wff2, dff, D);

    auto gemm = [&](const f16* A, const f16* WT_, const float* bias_,
                    const float* mul_, const float* res_,
                    float* o32, f16* o16, int Mx, int Nx, int Kx, int act) {
        dim3 grid((Nx + 127) / 128, (Mx + 127) / 128);
        gemm_wmma_kernel<<<grid, dim3(256), 0, stream>>>(A, WT_, bias_, mul_, res_,
                                                         o32, o16, Mx, Nx, Kx, act);
    };

    // stage 1: norms + conv
    cast16_kernel<<<(M * D + 255) / 256, 256, 0, stream>>>(x, x16, M * D);
    rmsnorm_kernel<<<M, 256, 0, stream>>>(x, nw, rms_tmp, (f16*)nullptr, D);
    conv_silu_kernel<<<M, 256, 0, stream>>>(rms_tmp, cw, cb, h16, T, D);

    // stage 2: projections (WMMA GEMMs)
    gemm(h16, WqT, nullptr, nullptr, nullptr, qf, nullptr, M, HD, D, 0);
    gemm(h16, WkT, nullptr, nullptr, nullptr, kf, nullptr, M, HD, D, 0);
    gemm(h16, WvT, nullptr, nullptr, nullptr, vf, nullptr, M, HD, D, 0);
    gemm(h16, WspT, nullptr, nullptr, nullptr, dr, nullptr, M, HD, D, 0);
    gemm(h16, WauT, bau, nullptr, nullptr, nullptr, u16, M, r, D, 1);      // silu
    gemm(u16, WadT, bad_, nullptr, nullptr, alb, nullptr, M, HD, r, 0);
    gemm(h16, WbT, bbeta, nullptr, nullptr, btb, nullptr, M, H, D, 0);

    l2norm_head_kernel<<<M, 512, 0, stream>>>(qf);
    l2norm_head_kernel<<<M, 512, 0, stream>>>(kf);

    // stage 3: sequential spiking delta-rule scan
    scan_kernel<<<Bb * H, 256, 0, stream>>>(qf, kf, vf, dr, alb, btb, asp, bsp, ysc, T, H);

    // stage 4: head norm + gated output projection + residual
    headrms_kernel<<<M, 512, 0, stream>>>(ysc, hnw, yn16);
    gemm(x16, Wu1T, bu1, nullptr, nullptr, nullptr, g16, M, r, D, 1);      // silu
    gemm(g16, Wu2T, bu2, nullptr, nullptr, gate, nullptr, M, D, r, 2);     // sigmoid
    gemm(yn16, WoT, nullptr, gate, x, y2, nullptr, M, D, HD, 0);           // *gate + x

    // stage 5: SwiGLU FFN + residual
    rmsnorm_kernel<<<M, 256, 0, stream>>>(y2, ffw, nullptr, z16, D);
    gemm(z16, Wff1T, nullptr, nullptr, nullptr, f1, nullptr, M, dff, D, 1);   // silu
    gemm(z16, Wff3T, nullptr, f1, nullptr, nullptr, p16, M, dff, D, 0);       // * silu branch
    gemm(p16, Wff2T, nullptr, nullptr, y2, out, nullptr, M, D, dff, 0);       // + y2
}
